// Receiver_76587856822495
// MI455X (gfx1250) — matrix operand order repaired and checked by reading, not compiled, not run on hardware
//
#include <hip/hip_runtime.h>
#include <hip/hip_bf16.h>

typedef __attribute__((ext_vector_type(16))) _Float16 v16h;
typedef __attribute__((ext_vector_type(8)))  _Float16 v8h;
typedef __attribute__((ext_vector_type(8)))  float    v8f;

#define HID 128
#define HC  128
#define NEG_SLOPE 0.2f

// ---------------- utility kernels ----------------

__global__ void k_fill(float* p, float v, long long n) {
  long long g = (long long)blockIdx.x * blockDim.x + threadIdx.x;
  long long stride = (long long)gridDim.x * blockDim.x;
  for (; g < n; g += stride) p[g] = v;
}

__global__ void k_reduce_sum(const float* in, int n, float* out_sum) {
  __shared__ float s[256];
  int t = threadIdx.x;
  float acc = 0.f;
  for (int i = blockIdx.x * 256 + t; i < n; i += gridDim.x * 256) acc += in[i];
  s[t] = acc; __syncthreads();
  for (int k = 128; k > 0; k >>= 1) { if (t < k) s[t] += s[t + k]; __syncthreads(); }
  if (t == 0) atomicAdd(out_sum, s[0]);
}

__global__ void k_finalize_mean(float* scal, float inv_n) {
  if (blockIdx.x == 0 && threadIdx.x == 0) scal[1] = scal[0] * inv_n;
}

__global__ void k_f32_to_f16(const float* in, _Float16* out, long long n) {
  long long g = (long long)blockIdx.x * blockDim.x + threadIdx.x;
  if (g < n) out[g] = (_Float16)in[g];
}

// in f32 [rows,cols] -> out f16 [cols,rows]  (builds transposed-B operands)
__global__ void k_transpose_f16(const float* in, _Float16* out, int rows, int cols) {
  long long g = (long long)blockIdx.x * blockDim.x + threadIdx.x;
  if (g >= (long long)rows * cols) return;
  int r = (int)(g / cols), c = (int)(g % cols);
  out[(long long)c * rows + r] = (_Float16)in[g];
}

__device__ __forceinline__ void atomicMaxFloat(float* addr, float val) {
  if (val >= 0.f) atomicMax((int*)addr, __float_as_int(val));
  else            atomicMin((unsigned int*)addr, __float_as_uint(val));
}

// ---------------- conv1 node transforms (rank-1) ----------------

__global__ void k_lin1(const float* x, const float* Wl, const float* bl,
                       const float* Wr, const float* br,
                       float* XL, float* XR, int N) {
  long long g = (long long)blockIdx.x * blockDim.x + threadIdx.x;
  if (g >= (long long)N * HC) return;
  int i = (int)(g >> 7), c = (int)(g & 127);
  float xv = x[i];
  XL[g] = xv * Wl[c] + bl[c];
  XR[g] = xv * Wr[c] + br[c];
}

// ---------------- edge passes (GATv2 attention) ----------------

__global__ void k_edge_alpha(const float* XL, const float* XR,
                             const float* edge_attr, const float* scal,
                             const float* We, const float* att,
                             const long long* ei, int E, int N,
                             float* ALPHA, float* SEGMAX) {
  int e = blockIdx.x * blockDim.x + threadIdx.x;
  int Et = E + N;
  if (e >= Et) return;
  long long src, tgt; float ea;
  if (e < E) { src = ei[e]; tgt = ei[E + e]; ea = edge_attr[e]; }
  else       { src = tgt = e - E;            ea = scal[1]; }
  const float* xl = XL + src * HC;
  const float* xr = XR + tgt * HC;
  #pragma unroll 2
  for (int h = 0; h < 2; ++h) {
    float a = 0.f;
    for (int c = 0; c < 64; ++c) {
      int idx = h * 64 + c;
      float m = xr[idx] + xl[idx] + ea * We[idx];
      m = (m > 0.f) ? m : m * NEG_SLOPE;
      a += m * att[idx];
    }
    ALPHA[(long long)e * 2 + h] = a;
    atomicMaxFloat(&SEGMAX[tgt * 2 + h], a);
  }
}

__global__ void k_edge_exp(float* ALPHA, const float* SEGMAX, float* DENOM,
                           const long long* ei, int E, int N) {
  long long g = (long long)blockIdx.x * blockDim.x + threadIdx.x;
  long long Et2 = 2LL * (E + N);
  if (g >= Et2) return;
  int e = (int)(g >> 1), h = (int)(g & 1);
  long long tgt = (e < E) ? ei[E + e] : (long long)(e - E);
  float ex = __expf(ALPHA[g] - SEGMAX[tgt * 2 + h]);
  ALPHA[g] = ex;
  atomicAdd(&DENOM[tgt * 2 + h], ex);
}

__global__ void k_edge_aggr(const float* XL, const float* ALPHA, const float* DENOM,
                            const long long* ei, int E, int N, float* ACC) {
  long long g = (long long)blockIdx.x * blockDim.x + threadIdx.x;
  long long Et4 = 4LL * (E + N);
  if (g >= Et4) return;
  int e = (int)(g >> 2), q = (int)(g & 3);
  long long src, tgt;
  if (e < E) { src = ei[e]; tgt = ei[E + e]; } else { src = tgt = e - E; }
  int h = q >> 1;
  float a = ALPHA[(long long)e * 2 + h] / DENOM[tgt * 2 + h];
  const float* xl = XL + src * HC + q * 32;
  float* acc = ACC + tgt * HC + q * 32;
  #pragma unroll
  for (int c = 0; c < 32; ++c) atomicAdd(&acc[c], xl[c] * a);
}

__global__ void k_relu_bias_h16(const float* ACC, const float* bias,
                                _Float16* H16, int N) {
  long long g = (long long)blockIdx.x * blockDim.x + threadIdx.x;
  if (g >= (long long)N * HC) return;
  int c = (int)(g & 127);
  float v = ACC[g] + bias[c];
  H16[g] = (_Float16)fmaxf(v, 0.f);
}

// ---------------- WMMA GEMM: C[M,Nc] = A[M,K]f16 x BT[Nc,K]f16^T (+bias) ----------------
// One wave per 16x16 C tile; K compile-time (full unroll -> 4 chained v_wmma).
// Operand layouts used here (A/B k order contiguous, same for both; C per ISA):
//   A lane l: row m0+(l&15); K offset 0 / 16 (contiguous, matches B)
//   B lane l: col n0+(l&15); K offset 0 / 16  -> contiguous in BT row-major
//   C VGPR r: row m0 + (l>=16 ? 8:0) + r, col n0+(l&15)
#define GWAVES 4

template <int K>
__global__ void k_gemm_wmma_bt(const _Float16* __restrict__ A,
                               const _Float16* __restrict__ BT,
                               const float* __restrict__ bias,
                               float* __restrict__ C,
                               int M, int Nc) {
  int lane = threadIdx.x & 31;
  int wave = threadIdx.x >> 5;
  int m0 = (blockIdx.y * GWAVES + wave) * 16;
  int n0 = blockIdx.x * 16;
  if (m0 >= M) return;                 // wave-uniform: EXEC stays all-ones

  int half_hi = lane >> 4;             // 0 | 1
  int mrow    = m0 + (lane & 15);
  int a_kbase = half_hi * 16;
  int b_koff  = half_hi * 16;
  int ncol    = n0 + (lane & 15);

  const _Float16* Arow = A  + (long long)mrow * K;
  const _Float16* Brow = BT + (long long)ncol * K;
  __builtin_prefetch(Arow, 0, 3);      // global_prefetch_b8
  __builtin_prefetch(Brow, 0, 3);

  v8f acc = {};
  #pragma unroll
  for (int k0 = 0; k0 < K; k0 += 32) {
    v16h av, bv;
    v8h alo = *(const v8h*)(Arow + k0 + a_kbase);        // 16B aligned
    v8h ahi = *(const v8h*)(Arow + k0 + 8 + a_kbase);
    #pragma unroll
    for (int j = 0; j < 8; ++j) { av[j] = alo[j]; av[8 + j] = ahi[j]; }
    bv = *(const v16h*)(Brow + k0 + b_koff);             // 32B aligned
    acc = __builtin_amdgcn_wmma_f32_16x16x32_f16(
        /*neg_a=*/false, av, /*neg_b=*/false, bv,
        /*c_mod=*/(short)0, acc, /*reuse_a=*/false, /*reuse_b=*/false);
  }

  int mo = m0 + half_hi * 8;
  float bb = bias ? bias[ncol] : 0.f;
  #pragma unroll
  for (int r = 0; r < 8; ++r)
    C[(long long)(mo + r) * Nc + ncol] = acc[r] + bb;
}

// ---------------- row softmax over Nc=256 ----------------

__global__ void k_softmax_rows(float* out, int Nc) {
  __shared__ float red[256];
  int row = blockIdx.x, t = threadIdx.x;
  float v = out[(long long)row * Nc + t];
  red[t] = v; __syncthreads();
  for (int s = 128; s > 0; s >>= 1) { if (t < s) red[t] = fmaxf(red[t], red[t + s]); __syncthreads(); }
  float mx = red[0]; __syncthreads();
  float e = __expf(v - mx);
  red[t] = e; __syncthreads();
  for (int s = 128; s > 0; s >>= 1) { if (t < s) red[t] += red[t + s]; __syncthreads(); }
  out[(long long)row * Nc + t] = e / red[0];
}

// ---------------- host orchestration ----------------

static inline int cdiv(long long a, long long b) { return (int)((a + b - 1) / b); }

extern "C" void kernel_launch(void* const* d_in, const int* in_sizes, int n_in,
                              void* d_out, int out_size, void* d_ws, size_t ws_size,
                              hipStream_t stream) {
  const float* message = (const float*)d_in[0];   // [B,128]
  const float* x       = (const float*)d_in[1];   // [N,1]
  const float* eattr   = (const float*)d_in[2];   // [E,1]
  const float* c1_Wl   = (const float*)d_in[3];
  const float* c1_bl   = (const float*)d_in[4];
  const float* c1_Wr   = (const float*)d_in[5];
  const float* c1_br   = (const float*)d_in[6];
  const float* c1_We   = (const float*)d_in[7];
  const float* c1_att  = (const float*)d_in[8];
  const float* c1_bias = (const float*)d_in[9];
  const float* c2_Wl   = (const float*)d_in[10];
  const float* c2_bl   = (const float*)d_in[11];
  const float* c2_Wr   = (const float*)d_in[12];
  const float* c2_br   = (const float*)d_in[13];
  const float* c2_We   = (const float*)d_in[14];
  const float* c2_att  = (const float*)d_in[15];
  const float* c2_bias = (const float*)d_in[16];
  const float* fc1_W   = (const float*)d_in[17];
  const float* fc1_b   = (const float*)d_in[18];
  const long long* ei  = (const long long*)d_in[19]; // [2,E] int64

  const int N  = in_sizes[1];
  const int E  = in_sizes[2];
  const int Bm = in_sizes[0] / HID;   // 256 candidate messages
  const int Et = E + N;

  // ---- carve workspace (256B aligned) ----
  char* base = (char*)d_ws;
  size_t off = 0;
  auto carve = [&](size_t bytes) -> char* {
    char* p = base + off;
    off = (off + bytes + 255) & ~(size_t)255;
    return p;
  };
  float*    SCAL   = (float*)   carve(2 * sizeof(float));            // [sum, mean]
  float*    XL     = (float*)   carve((size_t)N * HC * 4);
  float*    XR     = (float*)   carve((size_t)N * HC * 4);
  float*    ACC    = (float*)   carve((size_t)N * HC * 4);
  float*    ALPHA  = (float*)   carve((size_t)Et * 2 * 4);
  float*    SEGMAX = (float*)   carve((size_t)N * 2 * 4);
  float*    DENOM  = (float*)   carve((size_t)N * 2 * 4);
  float*    MSGEMB = (float*)   carve((size_t)Bm * HC * 4);
  _Float16* H16    = (_Float16*)carve((size_t)N * HC * 2);
  _Float16* W16A   = (_Float16*)carve((size_t)HC * HC * 2);   // transposed weights
  _Float16* W16B   = (_Float16*)carve((size_t)HC * HC * 2);
  _Float16* MSG16  = (_Float16*)carve((size_t)Bm * HID * 2);
  _Float16* ME16   = (_Float16*)carve((size_t)Bm * HC * 2);   // msg_emb f16 (row-major = BT for logits)

  const int T = 256;

  // ---- edge_attr mean (self-loop fill value) ----
  k_fill<<<1, T, 0, stream>>>(SCAL, 0.f, 2);
  k_reduce_sum<<<1024, T, 0, stream>>>(eattr, E, SCAL);
  k_finalize_mean<<<1, 64, 0, stream>>>(SCAL, 1.f / (float)E);

  // ================= conv1 =================
  k_lin1<<<cdiv((long long)N * HC, T), T, 0, stream>>>(x, c1_Wl, c1_bl, c1_Wr, c1_br, XL, XR, N);
  k_fill<<<2048, T, 0, stream>>>(SEGMAX, -__builtin_inff(), (long long)N * 2);
  k_fill<<<2048, T, 0, stream>>>(DENOM, 0.f, (long long)N * 2);
  k_fill<<<4096, T, 0, stream>>>(ACC, 0.f, (long long)N * HC);
  k_edge_alpha<<<cdiv(Et, T), T, 0, stream>>>(XL, XR, eattr, SCAL, c1_We, c1_att, ei, E, N, ALPHA, SEGMAX);
  k_edge_exp<<<cdiv(2LL * Et, T), T, 0, stream>>>(ALPHA, SEGMAX, DENOM, ei, E, N);
  k_edge_aggr<<<cdiv(4LL * Et, T), T, 0, stream>>>(XL, ALPHA, DENOM, ei, E, N, ACC);
  k_relu_bias_h16<<<cdiv((long long)N * HC, T), T, 0, stream>>>(ACC, c1_bias, H16, N);

  // ================= conv2 linear transforms via WMMA =================
  // W16A/W16B = W^T  (BT[n][k] = W[k][n])
  k_transpose_f16<<<cdiv(HC * HC, T), T, 0, stream>>>(c2_Wl, W16A, HC, HC);
  k_transpose_f16<<<cdiv(HC * HC, T), T, 0, stream>>>(c2_Wr, W16B, HC, HC);
  {
    dim3 grid(HC / 16, cdiv(N / 16, GWAVES));
    k_gemm_wmma_bt<HID><<<grid, 32 * GWAVES, 0, stream>>>(H16, W16A, c2_bl, XL, N, HC);
    k_gemm_wmma_bt<HID><<<grid, 32 * GWAVES, 0, stream>>>(H16, W16B, c2_br, XR, N, HC);
  }

  // ================= conv2 attention =================
  k_fill<<<2048, T, 0, stream>>>(SEGMAX, -__builtin_inff(), (long long)N * 2);
  k_fill<<<2048, T, 0, stream>>>(DENOM, 0.f, (long long)N * 2);
  k_fill<<<4096, T, 0, stream>>>(ACC, 0.f, (long long)N * HC);
  k_edge_alpha<<<cdiv(Et, T), T, 0, stream>>>(XL, XR, eattr, SCAL, c2_We, c2_att, ei, E, N, ALPHA, SEGMAX);
  k_edge_exp<<<cdiv(2LL * Et, T), T, 0, stream>>>(ALPHA, SEGMAX, DENOM, ei, E, N);
  k_edge_aggr<<<cdiv(4LL * Et, T), T, 0, stream>>>(XL, ALPHA, DENOM, ei, E, N, ACC);
  k_relu_bias_h16<<<cdiv((long long)N * HC, T), T, 0, stream>>>(ACC, c2_bias, H16, N);  // H16 = h2

  // ================= fc1: msg_emb = message @ fc1_W + fc1_b =================
  k_f32_to_f16<<<cdiv(Bm * HID, T), T, 0, stream>>>(message, MSG16, (long long)Bm * HID);
  k_transpose_f16<<<cdiv(HID * HC, T), T, 0, stream>>>(fc1_W, W16A, HID, HC);
  {
    dim3 grid(HC / 16, cdiv(Bm / 16, GWAVES));
    k_gemm_wmma_bt<HID><<<grid, 32 * GWAVES, 0, stream>>>(MSG16, W16A, fc1_b, MSGEMB, Bm, HC);
  }
  // logits BT = msg_emb itself (row-major [Bm, HC]) -> just convert to f16
  k_f32_to_f16<<<cdiv((long long)Bm * HC, T), T, 0, stream>>>(MSGEMB, ME16, (long long)Bm * HC);

  // ================= logits = h2 @ msg_emb^T, then row softmax =================
  {
    dim3 grid(Bm / 16, cdiv(N / 16, GWAVES));
    k_gemm_wmma_bt<HC><<<grid, 32 * GWAVES, 0, stream>>>(H16, ME16, nullptr, (float*)d_out, N, Bm);
  }
  k_softmax_rows<<<N, Bm, 0, stream>>>((float*)d_out, Bm);
}
